// SharedCodebook_24163486007567
// MI455X (gfx1250) — matrix operand order repaired and checked by reading, not compiled, not run on hardware
//
#include <hip/hip_runtime.h>
#include <hip/hip_bf16.h>
#include <math.h>
#include <stdint.h>

typedef __attribute__((ext_vector_type(16))) _Float16 v16h;
typedef __attribute__((ext_vector_type(8)))  _Float16 v8h;
typedef __attribute__((ext_vector_type(8)))  float    v8f;

#define BATCH   32768
#define NCODES  4096
#define DIM     1024
#define MTILE   32           // feature rows per block
#define NCHUNK  256          // codes per online-softmax step
#define QSTR    1032         // LDS row stride (halves) for Q tile: 16B-aligned + bank skew
#define SSTR    260          // LDS row stride (floats) for S tile: bank skew 4
#define PSTR    264          // LDS row stride (halves) for P tile: mult of 8 + bank skew

// ---------------------------------------------------------------------------
// Kernel 1: per-code-row L2 norm; emit normalized f16 row-major (B operand for
// Q·K^T, contiguous K=dim per lane) and raw f16 transposed [dim][code]
// (B operand for P·V, contiguous K=code per lane).
// ---------------------------------------------------------------------------
__global__ void __launch_bounds__(256)
codes_prep_kernel(const float* __restrict__ codes,
                  _Float16* __restrict__ codesn,   // [NCODES][DIM] normalized
                  _Float16* __restrict__ codesT)   // [DIM][NCODES] raw, transposed
{
    const int c = blockIdx.x;
    const int t = threadIdx.x;

    __shared__ float ssq;
    if (t == 0) ssq = 0.0f;
    __syncthreads();

    const float4 v = *(const float4*)(codes + (size_t)c * DIM + t * 4);
    float ps = v.x * v.x + v.y * v.y + v.z * v.z + v.w * v.w;
    #pragma unroll
    for (int off = 16; off > 0; off >>= 1) ps += __shfl_down(ps, off);
    if ((t & 31) == 0) atomicAdd(&ssq, ps);
    __syncthreads();

    const float inv = 1.0f / fmaxf(sqrtf(ssq), 1e-12f);

    _Float16* pn = codesn + (size_t)c * DIM + t * 4;
    pn[0] = (_Float16)(v.x * inv);
    pn[1] = (_Float16)(v.y * inv);
    pn[2] = (_Float16)(v.z * inv);
    pn[3] = (_Float16)(v.w * inv);

    const int d = t * 4;
    codesT[(size_t)(d + 0) * NCODES + c] = (_Float16)v.x;
    codesT[(size_t)(d + 1) * NCODES + c] = (_Float16)v.y;
    codesT[(size_t)(d + 2) * NCODES + c] = (_Float16)v.z;
    codesT[(size_t)(d + 3) * NCODES + c] = (_Float16)v.w;
}

// ---------------------------------------------------------------------------
// Kernel 2: fused  softmax(Fn·Cn^T)·codes  — flash-attention style.
// Block = 32 feature rows, 512 threads = 16 wave32s arranged 2 (M) x 8 (N/D).
// ---------------------------------------------------------------------------
__global__ void __launch_bounds__(512, 1)
codebook_attn_kernel(const float*    __restrict__ feats,
                     const _Float16* __restrict__ codesn,
                     const _Float16* __restrict__ codesT,
                     float*          __restrict__ out)
{
    // Persistent: normalized features f16 (66 KB)
    __shared__ __align__(16) _Float16 Qn[MTILE * QSTR];
    // Overlay region (128 KB): stage-A f32 feature staging OR S/P/reduction bufs
    __shared__ __align__(16) unsigned char ovl[MTILE * DIM * 4];
    float*    const Qstage  = (float*)ovl;                        // 131072 B (stage A)
    float*    const Sl      = (float*)ovl;                        // 33280 B
    _Float16* const Pl      = (_Float16*)(ovl + MTILE*SSTR*4);    // 16896 B
    float*    const partmax = (float*)(ovl + MTILE*SSTR*4 + MTILE*PSTR*2); // 2048 B
    __shared__ float mrow[MTILE], arow[MTILE], lrow[MTILE], rnorm[MTILE];

    const int t    = threadIdx.x;
    const int lane = t & 31;
    const int w    = t >> 5;            // wave 0..15
    const int mi   = w >> 3;            // M half: rows [mi*16, +16)
    const int wi   = w & 7;             // N/D slice id 0..7
    const int r0   = blockIdx.x * MTILE;

    const int nlo = lane & 15;          // B/C fragment column within tile
    const int hiA = (lane >> 4) * 8;    // C-fragment row offset
    const int hiB = (lane >> 4) * 16;   // B-fragment K sub-offset

    if (t < MTILE) { mrow[t] = -__builtin_inff(); lrow[t] = 0.0f; rnorm[t] = 0.0f; }
    __syncthreads();

    // ---- Stage A: async-DMA the 32x1024 f32 feature tile into LDS,
    //      then L2-normalize into Qn (f16).
    {
        const char*    gbase = (const char*)(feats + (size_t)r0 * DIM);
        const unsigned lbase = (unsigned)(uintptr_t)(void*)Qstage; // LDS byte offset
        #pragma unroll
        for (int i = 0; i < 16; ++i) {                 // 16 rounds x 8 KB/block
            const unsigned byte = (unsigned)(i * 8192 + t * 16);
            const unsigned ldst = lbase + byte;
            const char*    gsrc = gbase + byte;
            asm volatile("global_load_async_to_lds_b128 %0, %1, off"
                         :: "v"(ldst), "v"(gsrc) : "memory");
        }
#if __has_builtin(__builtin_amdgcn_s_wait_asynccnt)
        __builtin_amdgcn_s_wait_asynccnt(0);
#else
        asm volatile("s_wait_asynccnt 0" ::: "memory");
#endif
        __syncthreads();

        const int frow = t >> 4;                // 0..31
        const int fcol = (t & 15) * 64;
        const float4* fp = (const float4*)(Qstage + (size_t)frow * DIM + fcol);
        float4 fv[16];
        float  ps = 0.0f;
        #pragma unroll
        for (int i = 0; i < 16; ++i) {
            fv[i] = fp[i];
            ps += fv[i].x * fv[i].x + fv[i].y * fv[i].y +
                  fv[i].z * fv[i].z + fv[i].w * fv[i].w;
        }
        atomicAdd(&rnorm[frow], ps);
        __syncthreads();
        const float finv = 1.0f / fmaxf(sqrtf(rnorm[frow]), 1e-12f);
        #pragma unroll
        for (int j = 0; j < 8; ++j) {
            const float4 a = fv[2 * j], b = fv[2 * j + 1];
            v8h h = { (_Float16)(a.x * finv), (_Float16)(a.y * finv),
                      (_Float16)(a.z * finv), (_Float16)(a.w * finv),
                      (_Float16)(b.x * finv), (_Float16)(b.y * finv),
                      (_Float16)(b.z * finv), (_Float16)(b.w * finv) };
            *(v8h*)&Qn[frow * QSTR + fcol + 8 * j] = h;
        }
        __syncthreads();   // Qstage dead from here; overlay becomes S/P bufs
    }

    // ---- Output accumulators: wave owns rows [mi*16,+16) x D cols [wi*128,+128)
    v8f O[8];
    #pragma unroll
    for (int i = 0; i < 8; ++i) O[i] = (v8f){0.f,0.f,0.f,0.f,0.f,0.f,0.f,0.f};

    const int arow0 = (mi * 16 + nlo) * QSTR;
    const int prow0 = (mi * 16 + nlo) * PSTR;

    for (int chunk = 0; chunk < NCODES; chunk += NCHUNK) {
        if (chunk + NCHUNK < NCODES)       // pull next chunk's K rows toward L2
            __builtin_prefetch(codesn + (size_t)(chunk + NCHUNK + wi * 32 + nlo) * DIM, 0, 1);

        // ---- S = Qn · Kn^T : wave computes 16x32 (two n-tiles), A reused x2
        v8f S0 = (v8f){0.f,0.f,0.f,0.f,0.f,0.f,0.f,0.f};
        v8f S1 = (v8f){0.f,0.f,0.f,0.f,0.f,0.f,0.f,0.f};
        const _Float16* b0 = codesn + (size_t)(chunk + wi * 32 + nlo) * DIM + hiB;
        const _Float16* b1 = b0 + (size_t)16 * DIM;
        #pragma unroll
        for (int kk = 0; kk < 32; ++kk) {
            const int k0 = kk * 32;
            union { v16h v; v8h h[2]; } A;
            A.h[0] = *(const v8h*)&Qn[arow0 + k0 + hiB];
            A.h[1] = *(const v8h*)&Qn[arow0 + k0 + hiB + 8];
            const v16h B0 = *(const v16h*)(b0 + k0);
            const v16h B1 = *(const v16h*)(b1 + k0);
            S0 = __builtin_amdgcn_wmma_f32_16x16x32_f16(
                     false, A.v, false, B0, (short)0, S0, false, false);
            S1 = __builtin_amdgcn_wmma_f32_16x16x32_f16(
                     false, A.v, false, B1, (short)0, S1, false, false);
        }
        #pragma unroll
        for (int r = 0; r < 8; ++r) {
            const int row = mi * 16 + r + hiA;
            Sl[row * SSTR + wi * 32 + nlo]      = S0[r];
            Sl[row * SSTR + wi * 32 + 16 + nlo] = S1[r];
        }
        __syncthreads();

        // ---- parallel row-max: each thread scans 16 cols of one row
        {
            const int row = t >> 4, c0 = (t & 15) * 16;
            float mx = Sl[row * SSTR + c0];
            #pragma unroll
            for (int i = 1; i < 16; ++i) mx = fmaxf(mx, Sl[row * SSTR + c0 + i]);
            partmax[t] = mx;
        }
        __syncthreads();
        if (t < MTILE) {
            const float mo = mrow[t];
            float mx = mo;
            #pragma unroll
            for (int i = 0; i < 16; ++i) mx = fmaxf(mx, partmax[t * 16 + i]);
            const float al = __expf(mo - mx);
            mrow[t] = mx; arow[t] = al; lrow[t] *= al;
        }
        __syncthreads();

        // ---- P = exp(S - m) (f16), accumulate row sums; rescale O by alpha
        {
            const int row = t >> 4, c0 = (t & 15) * 16;
            const float m = mrow[row];
            float s = 0.0f;
            #pragma unroll
            for (int i = 0; i < 16; ++i) {
                const float p = __expf(Sl[row * SSTR + c0 + i] - m);
                s += p;
                Pl[row * PSTR + c0 + i] = (_Float16)p;
            }
            atomicAdd(&lrow[row], s);
        }
        #pragma unroll
        for (int r = 0; r < 8; ++r) {
            const float al = arow[mi * 16 + r + hiA];
            #pragma unroll
            for (int tt = 0; tt < 8; ++tt) O[tt][r] *= al;
        }
        __syncthreads();

        // ---- O += P · V : wave's 128-wide D slice, K = 256 codes (8 k-steps)
        #pragma unroll
        for (int kk = 0; kk < 8; ++kk) {
            const int k0 = kk * 32;
            union { v16h v; v8h h[2]; } A;
            A.h[0] = *(const v8h*)&Pl[prow0 + k0 + hiB];
            A.h[1] = *(const v8h*)&Pl[prow0 + k0 + hiB + 8];
            #pragma unroll
            for (int tt = 0; tt < 8; ++tt) {
                const _Float16* vp = codesT
                    + (size_t)(wi * 128 + tt * 16 + nlo) * NCODES + chunk + k0 + hiB;
                const v16h Bv = *(const v16h*)vp;
                O[tt] = __builtin_amdgcn_wmma_f32_16x16x32_f16(
                            false, A.v, false, Bv, (short)0, O[tt], false, false);
            }
        }
        __syncthreads();   // before S/P are overwritten next chunk
    }

    // ---- finalize: O / l, store f32
    float linv[8];
    #pragma unroll
    for (int r = 0; r < 8; ++r) linv[r] = 1.0f / lrow[mi * 16 + r + hiA];
    #pragma unroll
    for (int tt = 0; tt < 8; ++tt) {
        const int col = wi * 128 + tt * 16 + nlo;
        #pragma unroll
        for (int r = 0; r < 8; ++r) {
            const int row = r0 + mi * 16 + r + hiA;
            out[(size_t)row * DIM + col] = O[tt][r] * linv[r];
        }
    }
}

// ---------------------------------------------------------------------------
extern "C" void kernel_launch(void* const* d_in, const int* in_sizes, int n_in,
                              void* d_out, int out_size, void* d_ws, size_t ws_size,
                              hipStream_t stream) {
    const float* feats = (const float*)d_in[0];   // (32768, 1024) f32
    const float* codes = (const float*)d_in[1];   // (4096, 1024) f32
    float* out = (float*)d_out;                   // (32768, 1024) f32

    _Float16* codesn = (_Float16*)d_ws;                                   // 8 MB
    _Float16* codesT = (_Float16*)((char*)d_ws +
                        (size_t)NCODES * DIM * sizeof(_Float16));         // 8 MB

    hipLaunchKernelGGL(codes_prep_kernel, dim3(NCODES), dim3(256), 0, stream,
                       codes, codesn, codesT);
    hipLaunchKernelGGL(codebook_attn_kernel, dim3(BATCH / MTILE), dim3(512), 0, stream,
                       feats, codesn, codesT, out);
}
